// Qwen2MoELayer_64501818851347
// MI455X (gfx1250) — matrix operand order repaired and checked by reading, not compiled, not run on hardware
//
#include <hip/hip_runtime.h>
#include <hip/hip_bf16.h>

// ---------------- problem constants (from reference) ----------------
#define T_TOK 4096
#define HID   2048
#define NEXP  32
#define TOPK  4
#define FMOE  1408
#define FSH   5632
#define TK    (T_TOK * TOPK)        // 16384 expanded tokens
#define TILE_M 128
#define MAX_TILES 160               // TK/128 + NEXP upper bound

typedef __bf16 bf16;
typedef __attribute__((ext_vector_type(16))) __bf16 bf16x16;
typedef __attribute__((ext_vector_type(8)))  __bf16 bf16x8;
typedef __attribute__((ext_vector_type(8)))  float  f32x8;

union Frag { bf16x16 v; bf16x8 h[2]; };

// pack two fp32 -> packed bf16 pair (lowers to v_cvt_pk_bf16_f32)
__device__ __forceinline__ unsigned pk_bf16(float lo, float hi) {
  union { bf16 h[2]; unsigned u; } p;
  p.h[0] = (bf16)lo;
  p.h[1] = (bf16)hi;
  return p.u;
}

// ------- CDNA5 async global->LDS copy (ASYNCcnt tracked) -------------
// builtin prototype (from compiler diagnostic): (v4i as1*, v4i as3*, Ii, Ii)
typedef int v4i __attribute__((vector_size(16)));
#if __has_builtin(__builtin_amdgcn_global_load_async_to_lds_b128)
typedef __attribute__((address_space(1))) v4i g_v4i;
typedef __attribute__((address_space(3))) v4i l_v4i;
#define ASYNC_B128(gp, lp)                                                     \
  __builtin_amdgcn_global_load_async_to_lds_b128((g_v4i*)(gp), (l_v4i*)(lp),   \
                                                 0, 0)
#else
#define ASYNC_B128(gp, lp)                                                     \
  do {                                                                         \
    unsigned _lo = (unsigned)(unsigned long long)(const void*)(lp);            \
    asm volatile("global_load_async_to_lds_b128 %0, %1, off" ::"v"(_lo),       \
                 "v"((const void*)(gp))                                        \
                 : "memory");                                                  \
  } while (0)
#endif

#if __has_builtin(__builtin_amdgcn_s_wait_asynccnt)
#define WAIT_ASYNC() __builtin_amdgcn_s_wait_asynccnt(0)
#else
#define WAIT_ASYNC() asm volatile("s_wait_asynccnt 0x0" ::: "memory")
#endif

// =====================================================================
// 1) Router: logits -> softmax -> top-4.  One wave (32 lanes) per token,
//    lane e computes the dot-product with router_weight[e].
// =====================================================================
__global__ void router_topk_k(const float* __restrict__ hs,
                              const float* __restrict__ rweight,
                              float* __restrict__ rw, int* __restrict__ ridx) {
  int wave = threadIdx.x >> 5;
  int lane = threadIdx.x & 31;
  int t = blockIdx.x * 8 + wave;
  const float4* hp = (const float4*)(hs + (size_t)t * HID);
  const float4* wp = (const float4*)(rweight + (size_t)lane * HID);
  float acc = 0.f;
  #pragma unroll 8
  for (int i = 0; i < HID / 4; ++i) {
    float4 a = hp[i], b = wp[i];
    acc += a.x * b.x + a.y * b.y + a.z * b.z + a.w * b.w;
  }
  // softmax across the 32 lanes (= 32 experts)
  float m = acc;
  for (int s = 16; s; s >>= 1) m = fmaxf(m, __shfl_xor(m, s, 32));
  float e = __expf(acc - m);
  float sum = e;
  for (int s = 16; s; s >>= 1) sum += __shfl_xor(sum, s, 32);
  float p = e / sum;
  // iterative top-4 (ties -> lowest index, like lax.top_k)
  float pv = p;
  for (int j = 0; j < TOPK; ++j) {
    float v = pv; int ei = lane;
    for (int s = 16; s; s >>= 1) {
      float ov = __shfl_xor(v, s, 32);
      int   oe = __shfl_xor(ei, s, 32);
      if (ov > v || (ov == v && oe < ei)) { v = ov; ei = oe; }
    }
    if (lane == 0) { rw[t * TOPK + j] = v; ridx[t * TOPK + j] = ei; }
    if (lane == ei) pv = -1.f;
  }
}

// =====================================================================
// 2) Shared-expert sigmoid gate: gate[t] = sigmoid(h[t] . w)
// =====================================================================
__global__ void shared_gate_k(const float* __restrict__ hs,
                              const float* __restrict__ gw,
                              float* __restrict__ gate) {
  int wave = threadIdx.x >> 5;
  int lane = threadIdx.x & 31;
  int t = blockIdx.x * 8 + wave;
  const float4* hp = (const float4*)(hs + (size_t)t * HID);
  const float4* wp = (const float4*)gw;
  float acc = 0.f;
  for (int i = lane; i < HID / 4; i += 32) {
    float4 a = hp[i], b = wp[i];
    acc += a.x * b.x + a.y * b.y + a.z * b.z + a.w * b.w;
  }
  for (int s = 16; s; s >>= 1) acc += __shfl_xor(acc, s, 32);
  if (lane == 0) gate[t] = 1.f / (1.f + __expf(-acc));
}

// =====================================================================
// 3) Stable dispatch (single block): positions of each (token,k) in the
//    expert-sorted order, per-expert counts/offsets, and the M-tile map.
// =====================================================================
__global__ void dispatch_k(const int* __restrict__ ridx,
                           int* __restrict__ pos, int* __restrict__ r2t,
                           int* __restrict__ counts, int* __restrict__ offsets,
                           int* __restrict__ tiles) {
  __shared__ int sExp[256];
  __shared__ int sHist[NEXP];
  __shared__ int sBase[NEXP];
  __shared__ int sOff[NEXP];
  int tid = threadIdx.x;
  if (tid < NEXP) sBase[tid] = 0;
  __syncthreads();
  for (int c = 0; c < TK / 256; ++c) {
    int i = c * 256 + tid;
    int e = ridx[i];
    sExp[tid] = e;
    if (tid < NEXP) sHist[tid] = 0;
    __syncthreads();
    int rank = 0;
    for (int j = 0; j < tid; ++j) rank += (sExp[j] == e) ? 1 : 0;
    atomicAdd(&sHist[e], 1);
    pos[i] = sBase[e] + rank;      // expert-local position (stable)
    __syncthreads();
    if (tid < NEXP) sBase[tid] += sHist[tid];
    __syncthreads();
  }
  if (tid < NEXP) counts[tid] = sBase[tid];
  __syncthreads();
  if (tid == 0) {
    int run = 0, nt = 0;
    for (int e = 0; e < NEXP; ++e) {
      sOff[e] = run;
      offsets[e] = run;
      int cnt = sBase[e];
      int ntl = (cnt + TILE_M - 1) >> 7;       // 128-row tiles
      for (int j = 0; j < ntl; ++j) {
        tiles[1 + nt] = e;
        tiles[1 + MAX_TILES + nt] = run + j * TILE_M;
        ++nt;
      }
      run += cnt;
    }
    tiles[0] = nt;
  }
  __syncthreads();
  for (int i = tid; i < TK; i += 256) {
    int e = ridx[i];
    int p = pos[i] + sOff[e];
    pos[i] = p;                    // global sorted position
    r2t[p] = i >> 2;               // sorted row -> source token
  }
}

// =====================================================================
// 4) Unified 128x64-tile WMMA GEMM (bf16 in, fp32 accumulate).
//    8 waves; wave w owns rows [w*16, w*16+16) x all 64 columns
//    -> 4 (single) / 8 (dual) WMMA per 32-wide k-step.
//    DUAL:    B holds [gate | up]; epilogue = silu(g)*u (SwiGLU fuse)
//    A_BF16:  A staged via GLOBAL_LOAD_ASYNC_TO_LDS_B128 (ASYNCcnt)
//    GROUPED: per-expert GEMM driven by the device tile map
// =====================================================================
template <bool DUAL, bool A_BF16, bool GROUPED>
__global__ void gemm_wmma_k(const float* __restrict__ Afp,
                            const bf16* __restrict__ Abf,
                            const float* __restrict__ B,
                            bf16* __restrict__ Cout,
                            const int* __restrict__ r2t,
                            const int* __restrict__ tiles,
                            const int* __restrict__ offsets,
                            const int* __restrict__ counts,
                            int K, int N, int ldb, int rowsTotal) {
  __shared__ __align__(16) bf16 sA[TILE_M * 40];   // 10.0 KB
  __shared__ __align__(16) bf16 sB0[64 * 40];      //  5.0 KB
  __shared__ __align__(16) bf16 sB1[64 * 40];      //  5.0 KB

  int m0, rows_end;
  const float* Bexp;
  if (GROUPED) {
    int tile = blockIdx.x;
    if (tile >= tiles[0]) return;
    int e = tiles[1 + tile];
    m0 = tiles[1 + MAX_TILES + tile];
    rows_end = offsets[e] + counts[e];
    Bexp = B + (size_t)e * (size_t)K * (size_t)ldb;
  } else {
    m0 = blockIdx.x * TILE_M;
    rows_end = rowsTotal;
    Bexp = B;
  }
  int n0 = blockIdx.y * 64;

  int tid  = threadIdx.x;
  int wave = tid >> 5, lane = tid & 31;
  int l = lane & 15, hi = lane >> 4;

  // ---- precompute A staging addresses (k-invariant; gather hoisted) ----
  const float* arow[4];  bool avalidF[4]; int kkA = (tid & 7) * 4;
  const bf16*  abrow[2]; bool avalidB[2]; bf16* alds[2];
  if (A_BF16) {
    // 512 x b128 async transfers; 2 per thread
    #pragma unroll
    for (int i = 0; i < 2; ++i) {
      int lin = tid + i * 256;
      int r   = lin >> 2;               // 0..127
      int seg = (lin & 3) * 8;          // 0,8,16,24 bf16
      int grow = m0 + r;
      avalidB[i] = (grow < rows_end);
      abrow[i] = Abf + (size_t)(avalidB[i] ? grow : 0) * (size_t)K + seg;
      alds[i]  = &sA[r * 40 + seg];
      if (!avalidB[i]) *(uint4*)alds[i] = make_uint4(0u, 0u, 0u, 0u);
    }
  } else {
    // 1024 x float4 slots; 4 per thread
    #pragma unroll
    for (int i = 0; i < 4; ++i) {
      int r = (tid >> 3) + i * 32;      // 0..127
      int grow = m0 + r;
      avalidF[i] = (grow < rows_end);
      int srow = avalidF[i] ? (GROUPED ? r2t[grow] : grow) : 0;
      arow[i] = Afp + (size_t)srow * (size_t)K + kkA;
      if (!avalidF[i]) *(uint2*)&sA[r * 40 + kkA] = make_uint2(0u, 0u);
    }
  }

  f32x8 accG[4] = {{}, {}, {}, {}};
  f32x8 accU[4] = {{}, {}, {}, {}};

  for (int k0 = 0; k0 < K; k0 += 32) {
    // ---- stage A tile (128 rows x 32 k) as bf16 into LDS ----
    if (A_BF16) {
      #pragma unroll
      for (int i = 0; i < 2; ++i)
        if (avalidB[i]) ASYNC_B128(abrow[i] + k0, alds[i]);
    } else {
      #pragma unroll
      for (int i = 0; i < 4; ++i) {
        if (avalidF[i]) {
          int r = (tid >> 3) + i * 32;
          float4 av = *(const float4*)(arow[i] + k0);
          uint2 z;
          z.x = pk_bf16(av.x, av.y);
          z.y = pk_bf16(av.z, av.w);
          *(uint2*)&sA[r * 40 + kkA] = z;
        }
      }
    }
    // ---- stage B tile(s) (32 k x 64 n) transposed to [n][k] in LDS ----
    {
      int kp = tid >> 4;                  // 0..15 -> k rows 2kp, 2kp+1
      int c4 = (tid & 15) * 4;            // 0..60
      const float* bp0 = Bexp + (size_t)(k0 + 2 * kp) * (size_t)ldb + n0 + c4;
      const float* bp1 = bp0 + ldb;
      float4 g0 = *(const float4*)bp0;
      float4 g1 = *(const float4*)bp1;
      *(unsigned*)&sB0[(c4 + 0) * 40 + 2 * kp] = pk_bf16(g0.x, g1.x);
      *(unsigned*)&sB0[(c4 + 1) * 40 + 2 * kp] = pk_bf16(g0.y, g1.y);
      *(unsigned*)&sB0[(c4 + 2) * 40 + 2 * kp] = pk_bf16(g0.z, g1.z);
      *(unsigned*)&sB0[(c4 + 3) * 40 + 2 * kp] = pk_bf16(g0.w, g1.w);
      if (DUAL) {
        float4 u0 = *(const float4*)(bp0 + N);   // "up" columns live at +N
        float4 u1 = *(const float4*)(bp1 + N);
        *(unsigned*)&sB1[(c4 + 0) * 40 + 2 * kp] = pk_bf16(u0.x, u1.x);
        *(unsigned*)&sB1[(c4 + 1) * 40 + 2 * kp] = pk_bf16(u0.y, u1.y);
        *(unsigned*)&sB1[(c4 + 2) * 40 + 2 * kp] = pk_bf16(u0.z, u1.z);
        *(unsigned*)&sB1[(c4 + 3) * 40 + 2 * kp] = pk_bf16(u0.w, u1.w);
      }
    }
    if (A_BF16) WAIT_ASYNC();
    __syncthreads();

    // prefetch next k-step of B while we compute (global_prefetch_b8)
    if (k0 + 32 < K) {
      const float* pf =
          Bexp + (size_t)(k0 + 32 + (tid >> 3)) * (size_t)ldb + n0 + (tid & 7) * 8;
      __builtin_prefetch(pf, 0, 1);
      if (DUAL) __builtin_prefetch(pf + N, 0, 1);
    }

    // ---- fragments per the 16-bit 16x32 A / 32x16 B wave32 layouts ----
    Frag a;
    const bf16* abase = &sA[(wave * 16 + l) * 40 + hi * 8];
    a.h[0] = *(const bf16x8*)(abase);
    a.h[1] = *(const bf16x8*)(abase + 16);

    #pragma unroll
    for (int g = 0; g < 4; ++g) {
      Frag b;
      const bf16* bb = &sB0[(g * 16 + l) * 40 + hi * 8];
      b.h[0] = *(const bf16x8*)(bb);
      b.h[1] = *(const bf16x8*)(bb + 16);
      accG[g] = __builtin_amdgcn_wmma_f32_16x16x32_bf16(
          false, a.v, false, b.v, (short)0, accG[g], false, false);
      if (DUAL) {
        Frag u;
        const bf16* ub = &sB1[(g * 16 + l) * 40 + hi * 8];
        u.h[0] = *(const bf16x8*)(ub);
        u.h[1] = *(const bf16x8*)(ub + 16);
        accU[g] = __builtin_amdgcn_wmma_f32_16x16x32_bf16(
            false, a.v, false, u.v, (short)0, accU[g], false, false);
      }
    }
    __syncthreads();
  }

  // ---- epilogue: C/D layout -> acc[i] = (M = hi*8 + i, N = l) ----
  int row_lo = m0 + wave * 16 + hi * 8;
  #pragma unroll
  for (int g = 0; g < 4; ++g) {
    int col = n0 + g * 16 + l;
    #pragma unroll
    for (int i = 0; i < 8; ++i) {
      int grow = row_lo + i;
      if (grow >= rows_end) continue;
      float o;
      if (DUAL) {
        float gv = accG[g][i], uv = accU[g][i];
        o = (gv / (1.f + __expf(-gv))) * uv;   // silu(g)*u
      } else {
        o = accG[g][i];
      }
      Cout[(size_t)grow * (size_t)N + col] = (bf16)o;
    }
  }
}

// =====================================================================
// 5) Combine: out[t] = sum_k rw[t,k]*expert_out[pos[t,k]] + gate[t]*shared
// =====================================================================
__global__ void combine_k(const bf16* __restrict__ eout,
                          const bf16* __restrict__ sout,
                          const float* __restrict__ rw,
                          const int* __restrict__ pos,
                          const float* __restrict__ gate,
                          float* __restrict__ out) {
  int t = blockIdx.x;
  int p0 = pos[t * 4 + 0], p1 = pos[t * 4 + 1];
  int p2 = pos[t * 4 + 2], p3 = pos[t * 4 + 3];
  float w0 = rw[t * 4 + 0], w1 = rw[t * 4 + 1];
  float w2 = rw[t * 4 + 2], w3 = rw[t * 4 + 3];
  float g = gate[t];
  union V4 { uint2 u; bf16 h[4]; };
  for (int h4 = threadIdx.x * 4; h4 < HID; h4 += 256 * 4) {
    V4 a0, a1, a2, a3, s4;
    a0.u = *(const uint2*)(eout + (size_t)p0 * HID + h4);
    a1.u = *(const uint2*)(eout + (size_t)p1 * HID + h4);
    a2.u = *(const uint2*)(eout + (size_t)p2 * HID + h4);
    a3.u = *(const uint2*)(eout + (size_t)p3 * HID + h4);
    s4.u = *(const uint2*)(sout + (size_t)t  * HID + h4);
    float4 o;
    o.x = w0 * (float)a0.h[0] + w1 * (float)a1.h[0] + w2 * (float)a2.h[0] +
          w3 * (float)a3.h[0] + g * (float)s4.h[0];
    o.y = w0 * (float)a0.h[1] + w1 * (float)a1.h[1] + w2 * (float)a2.h[1] +
          w3 * (float)a3.h[1] + g * (float)s4.h[1];
    o.z = w0 * (float)a0.h[2] + w1 * (float)a1.h[2] + w2 * (float)a2.h[2] +
          w3 * (float)a3.h[2] + g * (float)s4.h[2];
    o.w = w0 * (float)a0.h[3] + w1 * (float)a1.h[3] + w2 * (float)a2.h[3] +
          w3 * (float)a3.h[3] + g * (float)s4.h[3];
    *(float4*)(out + (size_t)t * HID + h4) = o;
  }
}

// =====================================================================
// launcher
// =====================================================================
extern "C" void kernel_launch(void* const* d_in, const int* in_sizes, int n_in,
                              void* d_out, int out_size, void* d_ws, size_t ws_size,
                              hipStream_t stream) {
  const float* hidden  = (const float*)d_in[0];   // [T, H]
  const float* rweight = (const float*)d_in[1];   // [E, H]
  const float* gate_up = (const float*)d_in[2];   // [E, H, 2F]
  const float* downw   = (const float*)d_in[3];   // [E, F, H]
  const float* sh_gu   = (const float*)d_in[4];   // [H, 2FS]
  const float* sh_dn   = (const float*)d_in[5];   // [FS, H]
  const float* sh_gw   = (const float*)d_in[6];   // [H, 1]
  float* out = (float*)d_out;                     // [T, H] fp32

  // workspace carve-out (256B aligned slices)
  char* base = (char*)d_ws;
  size_t cur = 0;
  auto alloc = [&](size_t bytes) -> void* {
    void* p = base + cur;
    cur = (cur + bytes + 255) & ~(size_t)255;
    return p;
  };
  float* rw      = (float*)alloc((size_t)TK * 4);
  int*   ridx    = (int*)  alloc((size_t)TK * 4);
  int*   pos     = (int*)  alloc((size_t)TK * 4);
  int*   r2t     = (int*)  alloc((size_t)TK * 4);
  int*   counts  = (int*)  alloc((size_t)NEXP * 4);
  int*   offsets = (int*)  alloc((size_t)NEXP * 4);
  int*   tiles   = (int*)  alloc((size_t)(1 + 2 * MAX_TILES) * 4);
  float* gatev   = (float*)alloc((size_t)T_TOK * 4);
  bf16*  act     = (bf16*) alloc((size_t)TK * FMOE * 2);     // 46.1 MB
  bf16*  eout    = (bf16*) alloc((size_t)TK * HID * 2);      // 67.1 MB
  bf16*  sact    = (bf16*) alloc((size_t)T_TOK * FSH * 2);   // 46.1 MB
  bf16*  sout    = (bf16*) alloc((size_t)T_TOK * HID * 2);   // 16.8 MB

  // 1) router + shared gate
  router_topk_k<<<T_TOK / 8, 256, 0, stream>>>(hidden, rweight, rw, ridx);
  shared_gate_k<<<T_TOK / 8, 256, 0, stream>>>(hidden, sh_gw, gatev);

  // 2) stable dispatch + tile map
  dispatch_k<<<1, 256, 0, stream>>>(ridx, pos, r2t, counts, offsets, tiles);

  // 3) grouped expert GEMM1 (gathered A, fused SwiGLU) -> act[TK, F]
  gemm_wmma_k<true, false, true><<<dim3(MAX_TILES, FMOE / 64), 256, 0, stream>>>(
      hidden, nullptr, gate_up, act, r2t, tiles, offsets, counts,
      HID, FMOE, 2 * FMOE, TK);

  // 4) grouped expert GEMM2 (async bf16 A) -> expert_out[TK, H]
  gemm_wmma_k<false, true, true><<<dim3(MAX_TILES, HID / 64), 256, 0, stream>>>(
      nullptr, act, downw, eout, nullptr, tiles, offsets, counts,
      FMOE, HID, HID, TK);

  // 5) shared expert GEMM1 (fused SwiGLU) -> sact[T, FS]
  gemm_wmma_k<true, false, false><<<dim3(T_TOK / TILE_M, FSH / 64), 256, 0, stream>>>(
      hidden, nullptr, sh_gu, sact, nullptr, nullptr, nullptr, nullptr,
      HID, FSH, 2 * FSH, T_TOK);

  // 6) shared expert GEMM2 (async bf16 A) -> sout[T, H]
  gemm_wmma_k<false, true, false><<<dim3(T_TOK / TILE_M, HID / 64), 256, 0, stream>>>(
      nullptr, sact, sh_dn, sout, nullptr, nullptr, nullptr, nullptr,
      FSH, HID, HID, T_TOK);

  // 7) combine
  combine_k<<<T_TOK, 256, 0, stream>>>(eout, sout, rw, pos, gatev, out);

  (void)in_sizes; (void)n_in; (void)out_size; (void)ws_size;
}
